// AttentionD_12412455485977
// MI455X (gfx1250) — compile-verified, matrix operand order checked by reading, not run on hardware
//
#include <hip/hip_runtime.h>
#include <hip/hip_bf16.h>

typedef __attribute__((ext_vector_type(16))) _Float16     v16h;
typedef __attribute__((ext_vector_type(8)))  _Float16     v8h;
typedef __attribute__((ext_vector_type(8)))  float        v8f;
typedef __attribute__((ext_vector_type(4)))  unsigned int u32x4;
typedef __attribute__((ext_vector_type(8)))  unsigned int u32x8;

// K-offset (0..31) for element e (0..15) of an f16 A/B WMMA fragment:
// lanes 0-15:  VGPR0-3 -> K=0..7,  VGPR4-7 -> K=16..23
// lanes 16-31: VGPR0-3 -> K=8..15, VGPR4-7 -> K=24..31
__device__ __forceinline__ int ab_koff(int lane, int e) {
    int p = e >> 1, w = e & 1;
    return ((lane >> 4) ? 8 : 0) + ((p >= 4) ? 16 : 0) + ((p & 3) << 1) + w;
}

// Fragment gather when a lane's 16 halves live as two contiguous 16B runs at
// [gbase, gbase+8) and [gbase+16, gbase+24): two b128 loads + concat.
__device__ __forceinline__ v16h frag16(const _Float16* p, int gbase) {
    v8h lo = *(const v8h*)(p + gbase);
    v8h hi = *(const v8h*)(p + gbase + 16);
    return __builtin_shufflevector(lo, hi, 0, 1, 2, 3, 4, 5, 6, 7,
                                   8, 9, 10, 11, 12, 13, 14, 15);
}

// -------------------- Kernel 1: QKV GEMM --------------------
// x[4096,128] @ w_qkv[128,768] -> q,k f16 [b,h,n,dh]; v f16 TRANSPOSED [b,h,dh,n].
// q pre-scaled by dh^-0.5.
__global__ __launch_bounds__(128) void qkv_gemm_kernel(
    const float* __restrict__ x, const float* __restrict__ wqkv,
    _Float16* __restrict__ qb, _Float16* __restrict__ kb, _Float16* __restrict__ vb)
{
    const int lane = threadIdx.x;
    const int ln   = lane & 15;
    const int grp  = lane >> 4;
    const int m0   = (blockIdx.y * 4 + threadIdx.y) * 16;   // [0,4096)
    const int n0   = blockIdx.x * 16;                       // [0,768)
    const int mrow = m0 + ln;
    const int ncol = n0 + ln;

    v8f acc = {};
    for (int kc = 0; kc < 128; kc += 32) {
        v16h a, b;
#pragma unroll
        for (int e = 0; e < 16; ++e) {
            int ko = ab_koff(lane, e);
            a[e] = (_Float16)x[(long)mrow * 128 + kc + ko];
            b[e] = (_Float16)wqkv[(long)(kc + ko) * 768 + ncol];
        }
        acc = __builtin_amdgcn_wmma_f32_16x16x32_f16(
            false, a, false, b, (short)0, acc, false, false);
    }

    const float qscale = 0.17677669529663687f;  // 32^-0.5
    const int which = ncol >> 8;                // 0=q,1=k,2=v
    const int jj = ncol & 255;
    const int h = jj >> 5, dh = jj & 31;
#pragma unroll
    for (int r = 0; r < 8; ++r) {
        int m = m0 + r + 8 * grp;               // b*2048 + n
        int bidx = m >> 11, nrow = m & 2047;
        float val = acc[r];
        if (which == 0) {
            qb[(((long)(bidx * 8 + h)) * 2048 + nrow) * 32 + dh] =
                (_Float16)(val * qscale);
        } else if (which == 1) {
            kb[(((long)(bidx * 8 + h)) * 2048 + nrow) * 32 + dh] = (_Float16)val;
        } else {
            vb[(((long)(bidx * 8 + h)) * 32 + dh) * 2048 + nrow] = (_Float16)val;
        }
    }
}

// -------------------- Kernel 2: fused flash attention --------------------
// One block = one (b,h) x four 16-query tiles (one per wave). Per 32-key chunk:
// wave0 DMAs K (32x32 halves, row-major [key,dh]) and V^T (32x32, [dh,key])
// into LDS with tensor_load_to_lds (TENSORcnt), then all 4 waves run
// 4x v_wmma_f32_16x16x32_f16 + online softmax against the shared tiles.
__global__ __launch_bounds__(128) void flash_attn_kernel(
    const _Float16* __restrict__ qb, const _Float16* __restrict__ kb,
    const _Float16* __restrict__ vb, const float* __restrict__ rel_table,
    _Float16* __restrict__ ob)
{
    __shared__ __align__(16) _Float16 Kch[32 * 32];     // [key][dh]
    __shared__ __align__(16) _Float16 Vch[32 * 32];     // [dh][key]
    __shared__ __align__(16) _Float16 Ps[4][16 * 32];

    const int lane  = threadIdx.x;
    const int wid   = threadIdx.y;
    const int ln    = lane & 15;
    const int grp   = lane >> 4;
    const int gbase = grp ? 8 : 0;
    const int bh    = blockIdx.y;
    const int bidx  = bh >> 3, head = bh & 7;
    const int q0    = (blockIdx.x * 4 + wid) * 16;

    const _Float16* qp = qb + (long)bh * 2048 * 32;
    const unsigned long long kbase = (unsigned long long)(kb + (long)bh * 2048 * 32);
    const unsigned long long vbase = (unsigned long long)(vb + (long)bh * 32 * 2048);
    const unsigned ldsK = (unsigned)(uintptr_t)(&Kch[0]);
    const unsigned ldsV = (unsigned)(uintptr_t)(&Vch[0]);

    // Static D# group-1 descriptors (data_size=2B, 32x32 tiles).
    // K tensor: [2048 keys][32 dh], row stride 32.  V^T tensor: [32 dh][2048 keys].
    u32x8 kg1, vg1;
    kg1[0] = 0x00010000u;      // data_size = 2B
    kg1[1] = 32u << 16;        // tensor_dim0 = 32 (dh)
    kg1[2] = 2048u << 16;      // tensor_dim1 = 2048 (keys)
    kg1[3] = 32u << 16;        // tile_dim0 = 32
    kg1[4] = 32u;              // tile_dim1 = 32, tile_dim2 = 0
    kg1[5] = 32u;              // tensor_dim0_stride = 32
    kg1[6] = 0u;
    kg1[7] = 1u;               // tensor_dim1_stride = 65536
    vg1[0] = 0x00010000u;
    vg1[1] = 2048u << 16;      // tensor_dim0 = 2048 (keys)
    vg1[2] = 32u << 16;        // tensor_dim1 = 32 (dh)
    vg1[3] = 32u << 16;        // tile_dim0 = 32
    vg1[4] = 32u;              // tile_dim1 = 32
    vg1[5] = 2048u;            // tensor_dim0_stride = 2048
    vg1[6] = 0u;
    vg1[7] = 1u;

    // Q A-tile (already scaled): two global b128 loads per lane.
    v16h qa = frag16(qp + (long)(q0 + ln) * 32, gbase);

    float mrow[8], lrow[8];
    v8f o0 = {}, o1 = {};
#pragma unroll
    for (int r = 0; r < 8; ++r) { mrow[r] = -1e30f; lrow[r] = 0.0f; }

    for (int c0 = 0; c0 < 2048; c0 += 32) {
        if (wid == 0) {
            unsigned long long ka = kbase + (unsigned long long)c0 * 64ull;
            unsigned long long va = vbase + (unsigned long long)c0 * 2ull;
            u32x4 kg0, vg0;
            kg0[0] = 1u;                                   // count = 1
            kg0[1] = ldsK;                                 // lds_addr
            kg0[2] = (unsigned)(ka & 0xffffffffu);         // global_addr lo
            kg0[3] = (unsigned)((ka >> 32) & 0x01ffffffu) | (2u << 30); // hi | type=2
            vg0[0] = 1u;
            vg0[1] = ldsV;
            vg0[2] = (unsigned)(va & 0xffffffffu);
            vg0[3] = (unsigned)((va >> 32) & 0x01ffffffu) | (2u << 30);
            asm volatile("tensor_load_to_lds %0, %1" :: "s"(kg0), "s"(kg1) : "memory");
            asm volatile("tensor_load_to_lds %0, %1" :: "s"(vg0), "s"(vg1) : "memory");
            __builtin_amdgcn_s_wait_tensorcnt(0);
        }
        __syncthreads();

        // K B-tiles from LDS (2x ds_load_b128 each)
        v16h kt0 = frag16(&Kch[ln * 32], gbase);
        v16h kt1 = frag16(&Kch[(16 + ln) * 32], gbase);
        v8f zero = {};
        v8f d0 = __builtin_amdgcn_wmma_f32_16x16x32_f16(
            false, qa, false, kt0, (short)0, zero, false, false);
        v8f d1 = __builtin_amdgcn_wmma_f32_16x16x32_f16(
            false, qa, false, kt1, (short)0, zero, false, false);

        // relative position bias (analytic index; table lives in L2)
        const int j0 = c0 + ln, j1 = c0 + 16 + ln;
        const int j0z = j0 >> 8, j0y = (j0 >> 4) & 15, j0x = j0 & 15;
        const int j1z = j1 >> 8, j1y = (j1 >> 4) & 15, j1x = j1 & 15;
#pragma unroll
        for (int r = 0; r < 8; ++r) {
            int qi = q0 + r + 8 * grp;
            int qz = qi >> 8, qy = (qi >> 4) & 15, qx = qi & 15;
            int idx0 = (qz - j0z + 7) * 961 + (qy - j0y + 15) * 31 + (qx - j0x + 15);
            int idx1 = (qz - j1z + 7) * 961 + (qy - j1y + 15) * 31 + (qx - j1x + 15);
            d0[r] += rel_table[idx0 * 8 + head];
            d1[r] += rel_table[idx1 * 8 + head];
        }

        // online softmax over the 32-key chunk
        _Float16* ps = Ps[wid];
#pragma unroll
        for (int r = 0; r < 8; ++r) {
            float cm = fmaxf(d0[r], d1[r]);
#pragma unroll
            for (int mk = 1; mk < 16; mk <<= 1)
                cm = fmaxf(cm, __shfl_xor(cm, mk, 32));
            float nm = fmaxf(mrow[r], cm);
            float corr = __expf(mrow[r] - nm);
            float p0 = __expf(d0[r] - nm);
            float p1 = __expf(d1[r] - nm);
            float ls = p0 + p1;
#pragma unroll
            for (int mk = 1; mk < 16; mk <<= 1)
                ls += __shfl_xor(ls, mk, 32);
            lrow[r] = lrow[r] * corr + ls;
            mrow[r] = nm;
            o0[r] *= corr;
            o1[r] *= corr;
            ps[(r + 8 * grp) * 32 + ln]      = (_Float16)p0;
            ps[(r + 8 * grp) * 32 + 16 + ln] = (_Float16)p1;
        }
        asm volatile("s_wait_dscnt 0" ::: "memory");

        // P D-layout -> A-layout via per-wave LDS bounce
        v16h pa = frag16(&ps[ln * 32], gbase);
        // V B-tiles from transposed LDS tile: dh = ln (+16), key = K-dim
        v16h vt0 = frag16(&Vch[ln * 32], gbase);
        v16h vt1 = frag16(&Vch[(16 + ln) * 32], gbase);
        o0 = __builtin_amdgcn_wmma_f32_16x16x32_f16(
            false, pa, false, vt0, (short)0, o0, false, false);
        o1 = __builtin_amdgcn_wmma_f32_16x16x32_f16(
            false, pa, false, vt1, (short)0, o1, false, false);

        __syncthreads();   // protect Kch/Vch before next chunk's TDM overwrite
    }

    // normalize + write O as f16 [b, n, h*32+dh]
#pragma unroll
    for (int r = 0; r < 8; ++r) {
        float inv = 1.0f / lrow[r];
        int qi = q0 + r + 8 * grp;
        long row = (long)bidx * 2048 + qi;
        ob[row * 256 + head * 32 + ln]      = (_Float16)(o0[r] * inv);
        ob[row * 256 + head * 32 + 16 + ln] = (_Float16)(o1[r] * inv);
    }
}

// -------------------- Kernel 3: output projection --------------------
// O[4096,256] f16 @ w_out[256,128] (f32->f16) + b_out -> out f32
__global__ __launch_bounds__(128) void out_proj_kernel(
    const _Float16* __restrict__ ob, const float* __restrict__ wout,
    const float* __restrict__ bout, float* __restrict__ out)
{
    const int lane = threadIdx.x;
    const int ln   = lane & 15;
    const int grp  = lane >> 4;
    const int gbase = grp ? 8 : 0;
    const int m0   = (blockIdx.y * 4 + threadIdx.y) * 16;  // [0,4096)
    const int n0   = blockIdx.x * 16;                      // [0,128)

    v8f acc = {};
    for (int kc = 0; kc < 256; kc += 32) {
        v16h a = frag16(ob + (long)(m0 + ln) * 256 + kc, gbase);
        v16h b;
#pragma unroll
        for (int e = 0; e < 16; ++e) {
            int ko = ab_koff(lane, e);
            b[e] = (_Float16)wout[(long)(kc + ko) * 128 + n0 + ln];
        }
        acc = __builtin_amdgcn_wmma_f32_16x16x32_f16(
            false, a, false, b, (short)0, acc, false, false);
    }
    const int n = n0 + ln;
    const float bias = bout[n];
#pragma unroll
    for (int r = 0; r < 8; ++r) {
        int m = m0 + r + 8 * grp;
        out[(long)m * 128 + n] = acc[r] + bias;
    }
}

extern "C" void kernel_launch(void* const* d_in, const int* in_sizes, int n_in,
                              void* d_out, int out_size, void* d_ws, size_t ws_size,
                              hipStream_t stream) {
    const float* x    = (const float*)d_in[0];
    const float* wqkv = (const float*)d_in[1];
    const float* rel  = (const float*)d_in[2];
    const float* wout = (const float*)d_in[3];
    const float* bout = (const float*)d_in[4];
    // d_in[5..7] are d,h,w = 8,16,16 (hard-coded)

    _Float16* qb = (_Float16*)d_ws;      // 2 MB each
    _Float16* kb = qb + (1 << 20);
    _Float16* vb = kb + (1 << 20);       // transposed [b,h,dh,n]
    _Float16* ob = vb + (1 << 20);
    float* out = (float*)d_out;

    qkv_gemm_kernel<<<dim3(48, 64), dim3(32, 4), 0, stream>>>(x, wqkv, qb, kb, vb);
    flash_attn_kernel<<<dim3(32, 16), dim3(32, 4), 0, stream>>>(qb, kb, vb, rel, ob);
    out_proj_kernel<<<dim3(8, 64), dim3(32, 4), 0, stream>>>(ob, wout, bout, out);
}
